// GraphLearner_38019050504246
// MI455X (gfx1250) — matrix operand order repaired and checked by reading, not compiled, not run on hardware
//
#include <hip/hip_runtime.h>
#include <hip/hip_bf16.h>

// ---------------------------------------------------------------------------
// GraphLearner for gfx1250 (MI455X): bf16 WMMA GEMMs + exact radix top-k +
// in-place symmetrize/blend.  B=16, N=2048, D=512, H=4, k=614.
// ---------------------------------------------------------------------------

typedef __attribute__((ext_vector_type(16))) __bf16 v16bf;
typedef __attribute__((ext_vector_type(8)))  float  v8f;

#define BB   16
#define NN   2048
#define DD   512
#define KSEL 614
#define SIM_SCALE 0.02209708691207961f   // 1 / (sqrt(128) * 4)
#define LDSP 40                          // padded LDS row stride (elements)

// ---------------------------------------------------------------------------
// Kernel 1: Y = bf16(h @ W^T) for W in {Wq, Wk}   (M = B*N = 32768, K=N=512)
// Block: 256 thr (8 waves).  Block tile 64(M) x 128(Nout).
// Wave (w&3) -> 16-row strip, (w>>2) -> 64-col group = 4x wmma 16x16 tiles.
// ---------------------------------------------------------------------------
__global__ __launch_bounds__(256) void proj_kernel(
    const float* __restrict__ h,
    const float* __restrict__ Wq, const float* __restrict__ Wk,
    __bf16* __restrict__ Qb, __bf16* __restrict__ Kb)
{
    const float* W = blockIdx.z ? Wk : Wq;
    __bf16*      Y = blockIdx.z ? Kb : Qb;
    const int mBase = blockIdx.x * 64;
    const int nBase = blockIdx.y * 128;

    __shared__ float hT[64 * LDSP];    // 64 x 32 f32 tile of h
    __shared__ float wT[128 * LDSP];   // 128 x 32 f32 tile of W (rows = out cols)

    const int t    = threadIdx.x;
    const int lane = t & 31;
    const int wv   = t >> 5;
    const int mrow = (wv & 3) * 16;
    const int ngrp = (wv >> 2) * 64;
    const int lm   = lane & 15;
    const int hi   = lane >> 4;        // 0: lanes 0-15, 1: lanes 16-31

    v8f acc[4] = {};

    for (int k0 = 0; k0 < DD; k0 += 32) {
        // ---- stage h tile: 512 float4 slots, 2 per thread -----------------
        int f = t;
#pragma unroll
        for (int rep = 0; rep < 2; ++rep, f += 256) {
            int row = f >> 3, c4 = (f & 7) << 2;
            float4 v = *reinterpret_cast<const float4*>(
                &h[(size_t)(mBase + row) * DD + k0 + c4]);
            float* dst = &hT[row * LDSP + c4];
            dst[0] = v.x; dst[1] = v.y; dst[2] = v.z; dst[3] = v.w;
        }
        // ---- stage W tile: 1024 float4 slots, 4 per thread ----------------
        f = t;
#pragma unroll
        for (int rep = 0; rep < 4; ++rep, f += 256) {
            int row = f >> 3, c4 = (f & 7) << 2;
            float4 v = *reinterpret_cast<const float4*>(
                &W[(size_t)(nBase + row) * DD + k0 + c4]);
            float* dst = &wT[row * LDSP + c4];
            dst[0] = v.x; dst[1] = v.y; dst[2] = v.z; dst[3] = v.w;
        }
        if (k0 + 32 < DD) {  // prefetch next k-step (global_prefetch_b8)
            __builtin_prefetch(&h[(size_t)(mBase + (t >> 3)) * DD + k0 + 32], 0, 1);
            __builtin_prefetch(&W[(size_t)(nBase + (t >> 3)) * DD + k0 + 32], 0, 1);
        }
        __syncthreads();

        // ---- A fragment (16x32 bf16, contiguous k = 16*hi + s) -------------
        v16bf a;
        {
            const float* hrow = &hT[(mrow + lm) * LDSP];
            const int kb = hi * 16;
#pragma unroll
            for (int v = 0; v < 4; ++v) {
                a[2 * v]         = (__bf16)hrow[kb + 2 * v];
                a[2 * v + 1]     = (__bf16)hrow[kb + 2 * v + 1];
                a[8 + 2 * v]     = (__bf16)hrow[8 + kb + 2 * v];
                a[8 + 2 * v + 1] = (__bf16)hrow[8 + kb + 2 * v + 1];
            }
        }
        // ---- 4 B fragments + WMMA -----------------------------------------
#pragma unroll
        for (int tt = 0; tt < 4; ++tt) {
            const float* wrow = &wT[(ngrp + tt * 16 + lm) * LDSP + hi * 16];
            v16bf bf;
#pragma unroll
            for (int v = 0; v < 8; ++v) {
                bf[2 * v]     = (__bf16)wrow[2 * v];
                bf[2 * v + 1] = (__bf16)wrow[2 * v + 1];
            }
            acc[tt] = __builtin_amdgcn_wmma_f32_16x16x32_bf16(
                false, a, false, bf, (short)0, acc[tt], false, false);
        }
        __syncthreads();
    }

    // ---- store C (VGPR i -> rows i / i+8; N = lane%16) --------------------
#pragma unroll
    for (int tt = 0; tt < 4; ++tt) {
        const int col = nBase + ngrp + tt * 16 + lm;
#pragma unroll
        for (int i = 0; i < 8; ++i) {
            const int row = mBase + mrow + i + hi * 8;
            Y[(size_t)row * DD + col] = (__bf16)acc[tt][i];
        }
    }
}

// ---------------------------------------------------------------------------
// Kernel 2: sim[b] = SCALE * Q[b] @ K[b]^T  (per-batch 2048x2048x512, bf16 in)
// Same tiling; writes f32 sim into d_out (scratch reuse).
// ---------------------------------------------------------------------------
__global__ __launch_bounds__(256) void sim_kernel(
    const __bf16* __restrict__ Q, const __bf16* __restrict__ Km,
    float* __restrict__ sim)
{
    const int b = blockIdx.z;
    const int rBase = blockIdx.x * 64;    // n (rows of sim)
    const int cBase = blockIdx.y * 128;   // m (cols of sim)
    const __bf16* Qb = Q  + (size_t)b * NN * DD;
    const __bf16* Kb = Km + (size_t)b * NN * DD;

    __shared__ __bf16 qT[64 * LDSP];
    __shared__ __bf16 kT[128 * LDSP];

    const int t    = threadIdx.x;
    const int lane = t & 31;
    const int wv   = t >> 5;
    const int mrow = (wv & 3) * 16;
    const int ngrp = (wv >> 2) * 64;
    const int lm   = lane & 15;
    const int hi   = lane >> 4;

    v8f acc[4] = {};

    for (int k0 = 0; k0 < DD; k0 += 32) {
        // qT: 64 rows x 4 chunks of 8 bf16 = 256 chunks, 1/thread
        {
            int row = t >> 2, c8 = (t & 3) << 3;
            *reinterpret_cast<uint4*>(&qT[row * LDSP + c8]) =
                *reinterpret_cast<const uint4*>(&Qb[(size_t)(rBase + row) * DD + k0 + c8]);
        }
        // kT: 128 rows x 4 chunks = 512 chunks, 2/thread
        int f = t;
#pragma unroll
        for (int rep = 0; rep < 2; ++rep, f += 256) {
            int row = f >> 2, c8 = (f & 3) << 3;
            *reinterpret_cast<uint4*>(&kT[row * LDSP + c8]) =
                *reinterpret_cast<const uint4*>(&Kb[(size_t)(cBase + row) * DD + k0 + c8]);
        }
        if (k0 + 32 < DD) {
            __builtin_prefetch(&Qb[(size_t)(rBase + (t >> 2)) * DD + k0 + 32], 0, 1);
            __builtin_prefetch(&Kb[(size_t)(cBase + (t >> 2)) * DD + k0 + 32], 0, 1);
        }
        __syncthreads();

        v16bf a;
        {
            const __bf16* qrow = &qT[(mrow + lm) * LDSP];
            const int kb = hi * 16;
#pragma unroll
            for (int v = 0; v < 4; ++v) {
                a[2 * v]         = qrow[kb + 2 * v];
                a[2 * v + 1]     = qrow[kb + 2 * v + 1];
                a[8 + 2 * v]     = qrow[8 + kb + 2 * v];
                a[8 + 2 * v + 1] = qrow[8 + kb + 2 * v + 1];
            }
        }
#pragma unroll
        for (int tt = 0; tt < 4; ++tt) {
            const __bf16* krow = &kT[(ngrp + tt * 16 + lm) * LDSP + hi * 16];
            v16bf bf;
#pragma unroll
            for (int v = 0; v < 8; ++v) {
                bf[2 * v]     = krow[2 * v];
                bf[2 * v + 1] = krow[2 * v + 1];
            }
            acc[tt] = __builtin_amdgcn_wmma_f32_16x16x32_bf16(
                false, a, false, bf, (short)0, acc[tt], false, false);
        }
        __syncthreads();
    }

    float* out = sim + (size_t)b * NN * NN;
#pragma unroll
    for (int tt = 0; tt < 4; ++tt) {
        const int col = cBase + ngrp + tt * 16 + lm;
#pragma unroll
        for (int i = 0; i < 8; ++i) {
            const int row = rBase + mrow + i + hi * 8;
            out[(size_t)row * NN + col] = acc[tt][i] * SIM_SCALE;
        }
    }
}

// ---------------------------------------------------------------------------
// Kernel 3: exact k-th largest per row via 4-pass MSB radix select in LDS.
// One 256-thread block per (b,n) row; 8 elements/thread held in registers.
// ---------------------------------------------------------------------------
__global__ __launch_bounds__(256) void select_kernel(
    const float* __restrict__ sim, float* __restrict__ thresh)
{
    const int row = blockIdx.x;                 // 0 .. B*N-1
    const float* src = sim + (size_t)row * NN;
    const int t = threadIdx.x;

    // monotonic float -> u32 key (ascending)
    unsigned key[8];
#pragma unroll
    for (int i = 0; i < 8; ++i) {
        unsigned u = __float_as_uint(src[t + 256 * i]);
        key[i] = (u & 0x80000000u) ? ~u : (u | 0x80000000u);
    }

    __shared__ unsigned hist[256];
    __shared__ unsigned s_prefix, s_rank;
    if (t == 0) { s_prefix = 0u; s_rank = (unsigned)KSEL; }

    unsigned mask = 0u;
    for (int shift = 24; shift >= 0; shift -= 8) {
        hist[t] = 0u;
        __syncthreads();
        const unsigned pfx = s_prefix;
#pragma unroll
        for (int i = 0; i < 8; ++i)
            if ((key[i] & mask) == pfx)
                atomicAdd(&hist[(key[i] >> shift) & 255u], 1u);
        __syncthreads();
        if (t == 0) {
            unsigned r = s_rank, cum = 0u, bin = 0u;
            for (int b2 = 255; b2 >= 0; --b2) {
                unsigned c = hist[b2];
                if (cum + c >= r) { bin = (unsigned)b2; break; }
                cum += c;
            }
            s_rank   = r - cum;
            s_prefix = pfx | (bin << shift);
        }
        __syncthreads();
        mask |= (0xFFu << shift);
    }

    if (t == 0) {
        const unsigned kth = s_prefix;
        const unsigned u = (kth & 0x80000000u) ? (kth ^ 0x80000000u) : ~kth;
        thresh[row] = __uint_as_float(u);
    }
}

// ---------------------------------------------------------------------------
// Kernel 4: in-place mask/sigmoid/symmetrize/blend over 64x64 tile pairs.
// Block owns tiles (ti,tj) and (tj,ti); stage both to LDS, barrier, rewrite.
// ---------------------------------------------------------------------------
__global__ __launch_bounds__(256) void blend_kernel(
    float* __restrict__ sim_out, const float* __restrict__ adj,
    const float* __restrict__ thresh, const float* __restrict__ gamma_raw)
{
    const int b = blockIdx.y;
    int p = blockIdx.x;
    int ti = 0;
    { const int T = NN / 64; while (p >= T - ti) { p -= T - ti; ++ti; } }
    const int tj = ti + p;
    const int r0 = ti * 64, c0 = tj * 64;

    __shared__ float s1[64 * 65];
    __shared__ float s2[64 * 65];
    __shared__ float thA[64], thB[64];

    const int t = threadIdx.x;
    float* base = sim_out + (size_t)b * NN * NN;
    const float* adjb = adj + (size_t)b * NN * NN;

#pragma unroll
    for (int i = 0; i < 16; ++i) {
        const int idx = t + 256 * i;
        const int r = idx >> 6, c = idx & 63;
        s1[r * 65 + c] = base[(size_t)(r0 + r) * NN + c0 + c];
    }
    if (ti != tj) {
#pragma unroll
        for (int i = 0; i < 16; ++i) {
            const int idx = t + 256 * i;
            const int r = idx >> 6, c = idx & 63;
            s2[r * 65 + c] = base[(size_t)(c0 + r) * NN + r0 + c];
        }
    }
    if (t < 64) {
        thA[t] = thresh[b * NN + r0 + t];
        thB[t] = thresh[b * NN + c0 + t];
    }
    __syncthreads();

    const float g   = 1.0f / (1.0f + __expf(-gamma_raw[0]));
    const float omg = 1.0f - g;

#pragma unroll
    for (int i = 0; i < 16; ++i) {
        const int idx = t + 256 * i;
        const int r = idx >> 6, c = idx & 63;
        const float sij = s1[r * 65 + c];
        const float sji = (ti == tj) ? s1[c * 65 + r] : s2[c * 65 + r];
        const float aij = (sij >= thA[r]) ? 1.0f / (1.0f + __expf(-sij)) : 0.0f;
        const float aji = (sji >= thB[c]) ? 1.0f / (1.0f + __expf(-sji)) : 0.0f;
        const float sym = 0.5f * (aij + aji);
        const size_t gij = (size_t)(r0 + r) * NN + (c0 + c);
        base[gij] = omg * adjb[gij] + g * sym;
        if (ti != tj) {
            const size_t gji = (size_t)(c0 + c) * NN + (r0 + r);
            base[gji] = omg * adjb[gji] + g * sym;
        }
    }
}

// ---------------------------------------------------------------------------
extern "C" void kernel_launch(void* const* d_in, const int* in_sizes, int n_in,
                              void* d_out, int out_size, void* d_ws, size_t ws_size,
                              hipStream_t stream) {
    const float* h    = (const float*)d_in[0];
    const float* adj  = (const float*)d_in[1];
    const float* Wq   = (const float*)d_in[2];
    const float* Wk   = (const float*)d_in[3];
    const float* graw = (const float*)d_in[4];
    float* out = (float*)d_out;

    char* ws = (char*)d_ws;
    const size_t qkBytes = (size_t)BB * NN * DD * 2;     // 33,554,432 each
    __bf16* Qb     = (__bf16*)ws;
    __bf16* Kb     = (__bf16*)(ws + qkBytes);
    float*  thresh = (float*)(ws + 2 * qkBytes);         // B*N floats

    dim3 blk(256);
    // 1) projections
    dim3 g1((BB * NN) / 64, DD / 128, 2);
    proj_kernel<<<g1, blk, 0, stream>>>(h, Wq, Wk, Qb, Kb);
    // 2) scaled Q K^T -> sim in d_out
    dim3 g2(NN / 64, NN / 128, BB);
    sim_kernel<<<g2, blk, 0, stream>>>(Qb, Kb, out);
    // 3) per-row exact top-k threshold
    dim3 g3(BB * NN);
    select_kernel<<<g3, blk, 0, stream>>>(out, thresh);
    // 4) in-place mask + symmetrize + blend
    dim3 g4((NN / 64) * (NN / 64 + 1) / 2, BB);
    blend_kernel<<<g4, blk, 0, stream>>>(out, adj, thresh, graw);
}
